// PointNet2UnetForFlowMatching_58798102282446
// MI455X (gfx1250) — matrix operand order repaired and checked by reading, not compiled, not run on hardware
//
#include <hip/hip_runtime.h>
#include <hip/hip_bf16.h>
#include <math.h>

// PointNet++ UNet for flow matching — CDNA5 (gfx1250) implementation.
// Conv-chain GEMMs: f32->f16 pre-packed operands, v_wmma_f32_16x16x32_f16,
// fp32 accumulate, branch-free inner loop, 2x M-tiled per wave (B-fragment
// reuse -> ~10.7 FLOP/byte of global traffic).
// B=8, N=2048, TDIM=128 fixed by the reference.

#define BB 8
#define NNPTS 2048
#define TD 128

typedef __attribute__((ext_vector_type(16))) _Float16 v16h;
typedef __attribute__((ext_vector_type(8)))  _Float16 v8h;
typedef __attribute__((ext_vector_type(8)))  float    v8f;

__device__ __forceinline__ float gelu_f(float x) {
    // exact gelu: 0.5*x*(1+erf(x/sqrt(2)))
    return 0.5f * x * (1.0f + erff(x * 0.70710678118654752f));
}

// ---------------- time embedding ----------------
__global__ void k_time_sincos(const float* __restrict__ t, float* __restrict__ emb) {
    int i = blockIdx.x * blockDim.x + threadIdx.x;
    if (i >= BB * TD) return;
    int b = i / TD, d = i % TD;
    const int half = TD / 2;
    int j = (d < half) ? d : (d - half);
    float f = expf(-logf(10000.0f) * (float)j / (float)(half - 1));
    float e = t[b] * f;
    emb[i] = (d < half) ? sinf(e) : cosf(e);
}

__global__ void k_time_proj(const float* __restrict__ emb, const float* __restrict__ w,
                            const float* __restrict__ bias, float* __restrict__ te) {
    int i = blockIdx.x * blockDim.x + threadIdx.x;
    if (i >= BB * TD) return;
    int b = i / TD, o = i % TD;
    float s = bias[o];
    for (int k = 0; k < TD; ++k) s += emb[b * TD + k] * w[o * TD + k];
    te[i] = gelu_f(s);
}

// tproj[b,c] = tm_b[c] + sum_t gelu(te[b,t]) * tm_w[c,t]
__global__ void k_tproj(const float* __restrict__ te, const float* __restrict__ tmw,
                        const float* __restrict__ tmb, float* __restrict__ tp, int Cin) {
    int i = blockIdx.x * blockDim.x + threadIdx.x;
    if (i >= BB * Cin) return;
    int b = i / Cin, c = i % Cin;
    float s = tmb[c];
    for (int k = 0; k < TD; ++k) s += gelu_f(te[b * TD + k]) * tmw[c * TD + k];
    tp[i] = s;
}

// bias2[b,o] = conv_b[o] + sum_c W[o,c] * tproj[b,c]   (folds time conditioning into GEMM bias)
__global__ void k_bias2(const float* __restrict__ W, const float* __restrict__ cb,
                        const float* __restrict__ tp, float* __restrict__ b2, int Cin, int Cout) {
    int i = blockIdx.x * blockDim.x + threadIdx.x;
    if (i >= BB * Cout) return;
    int b = i / Cout, o = i % Cout;
    const float* wr = W  + (size_t)o * Cin;
    const float* tb = tp + (size_t)b * Cin;
    float s = cb[o];
    for (int c = 0; c < Cin; ++c) s += wr[c] * tb[c];
    b2[i] = s;
}

// ---------------- operand packing (f32 -> zero-padded f16) ----------------
// Wh (Cout, Cpad) row-major
__global__ void k_pack_w(const float* __restrict__ W, _Float16* __restrict__ Wh,
                         int Cin, int Cpad, int Cout) {
    size_t i = (size_t)blockIdx.x * blockDim.x + threadIdx.x;
    size_t total = (size_t)Cout * Cpad;
    if (i >= total) return;
    int k = (int)(i % Cpad);
    int o = (int)(i / Cpad);
    Wh[i] = (_Float16)((k < Cin) ? W[(size_t)o * Cin + k] : 0.0f);
}

// Xh (B, L, Cpad): K-contiguous so B fragments are b128 loads
__global__ void k_pack_x(const float* __restrict__ X, _Float16* __restrict__ Xh,
                         int Cin, int Cpad, int L) {
    size_t i = (size_t)blockIdx.x * blockDim.x + threadIdx.x;
    size_t total = (size_t)BB * L * Cpad;
    if (i >= total) return;
    int k = (int)(i % Cpad);
    size_t r = i / Cpad;
    int l = (int)(r % L);
    int b = (int)(r / L);
    float v = (k < Cin) ? X[((size_t)b * Cin + k) * L + l] : 0.0f;
    Xh[i] = (_Float16)v;
}

// ---------------- WMMA GEMM ----------------
// Y[b, o, l] = sum_k Wh[o,k]*Xh[b,l,k] + bias[b*bias_stride + o]
// 64-thread block = 2 waves; each wave owns TWO 16x16 tiles stacked in M
// (32 output rows), sharing one B fragment per K-step:
//   6x global_load_b128 -> 2x v_wmma per 32-wide K step.
__global__ void k_gemm_wmma(const _Float16* __restrict__ Wh, const _Float16* __restrict__ Xh,
                            const float* __restrict__ bias, float* __restrict__ Y,
                            int Cout, int Cpad, int L, int bias_stride) {
    int lt   = blockIdx.x;                           // L tile
    int wave = threadIdx.x >> 5;                     // 0..1
    int ot   = blockIdx.y * 4 + wave * 2;            // first of two M tiles
    int b    = blockIdx.z;
    int lane = threadIdx.x & 31;
    int hi   = lane >> 4;
    int lo   = lane & 15;

    // A fragment (16x32 f16): row m=lo; h<8 -> K=k0+8*hi+h ; h>=8 -> K=k0+16+8*hi+(h&7)
    const _Float16* ap0 = Wh + (size_t)(ot * 16 + lo) * Cpad + 8 * hi;
    const _Float16* ap1 = ap0 + (size_t)16 * Cpad;
    // B fragment (32x16 f16): col n=lo; h<8 -> K=k0+8*hi+h ; h>=8 -> K=k0+16+8*hi+(h&7)
    const _Float16* bp  = Xh + ((size_t)b * L + (size_t)lt * 16 + lo) * Cpad + 8 * hi;

    v8f acc0 = {};
    v8f acc1 = {};
#pragma unroll 2
    for (int k0 = 0; k0 < Cpad; k0 += 32) {
        v8h b0 = *(const v8h*)(bp + k0);
        v8h b1 = *(const v8h*)(bp + k0 + 16);
        v8h a00 = *(const v8h*)(ap0 + k0);
        v8h a01 = *(const v8h*)(ap0 + k0 + 16);
        v8h a10 = *(const v8h*)(ap1 + k0);
        v8h a11 = *(const v8h*)(ap1 + k0 + 16);
        v16h bf  = __builtin_shufflevector(b0, b1, 0, 1, 2, 3, 4, 5, 6, 7,
                                                   8, 9, 10, 11, 12, 13, 14, 15);
        v16h af0 = __builtin_shufflevector(a00, a01, 0, 1, 2, 3, 4, 5, 6, 7,
                                                     8, 9, 10, 11, 12, 13, 14, 15);
        v16h af1 = __builtin_shufflevector(a10, a11, 0, 1, 2, 3, 4, 5, 6, 7,
                                                     8, 9, 10, 11, 12, 13, 14, 15);
        acc0 = __builtin_amdgcn_wmma_f32_16x16x32_f16(
            false, af0, false, bf, (short)0, acc0, false, false);
        acc1 = __builtin_amdgcn_wmma_f32_16x16x32_f16(
            false, af1, false, bf, (short)0, acc1, false, false);
    }

    // C/D layout: VGPR r, lane -> m = r + 8*hi, n = lo
    const float* bb = bias + (size_t)b * bias_stride;
    float* Yb0 = Y + ((size_t)b * Cout + (size_t)ot * 16) * L + (size_t)lt * 16;
    float* Yb1 = Yb0 + (size_t)16 * L;
#pragma unroll
    for (int r = 0; r < 8; ++r) {
        int m = r + 8 * hi;
        Yb0[(size_t)m * L + lo] = acc0[r] + bb[ot * 16 + m];
        Yb1[(size_t)m * L + lo] = acc1[r] + bb[ot * 16 + 16 + m];
    }
}

// ---------------- batch-norm (stats over batch & length) + gelu ----------------
__global__ void k_bn_stats(const float* __restrict__ X, float* __restrict__ mean,
                           float* __restrict__ rstd, int C, int L) {
    __shared__ float ssum[256];
    __shared__ float ssq[256];
    int c = blockIdx.x, tid = threadIdx.x;
    float s = 0.0f, q = 0.0f;
    int total = BB * L;
    for (int i = tid; i < total; i += 256) {
        int b = i / L, l = i % L;
        float v = X[((size_t)b * C + c) * L + l];
        s += v; q += v * v;
    }
    ssum[tid] = s; ssq[tid] = q;
    __syncthreads();
    for (int off = 128; off > 0; off >>= 1) {
        if (tid < off) { ssum[tid] += ssum[tid + off]; ssq[tid] += ssq[tid + off]; }
        __syncthreads();
    }
    if (tid == 0) {
        float m = ssum[0] / (float)total;
        float v = ssq[0] / (float)total - m * m;
        mean[c] = m;
        rstd[c] = rsqrtf(v + 1e-5f);
    }
}

__global__ void k_bn_gelu(float* __restrict__ X, const float* __restrict__ mean,
                          const float* __restrict__ rstd, const float* __restrict__ gamma,
                          const float* __restrict__ beta, int C, int L, size_t total) {
    size_t i = (size_t)blockIdx.x * blockDim.x + threadIdx.x;
    if (i >= total) return;
    int c = (int)((i / (size_t)L) % (size_t)C);
    float v = X[i];
    v = (v - mean[c]) * rstd[c] * gamma[c] + beta[c];
    X[i] = gelu_f(v);
}

// ---------------- max pool over K ----------------
__global__ void k_maxpool(const float* __restrict__ X, float* __restrict__ Y,
                          int C, int S, int K) {
    size_t i = (size_t)blockIdx.x * blockDim.x + threadIdx.x;
    size_t total = (size_t)BB * C * S;
    if (i >= total) return;
    size_t base = i * (size_t)K;
    float m = X[base];
    for (int k = 1; k < K; ++k) m = fmaxf(m, X[base + k]);
    Y[i] = m;
}

// ---------------- farthest point sampling ----------------
__global__ void k_fps(const float* __restrict__ xyz, int* __restrict__ out, int N, int npoint) {
    __shared__ float dist[NNPTS];
    __shared__ float rmax[256];
    __shared__ int   rarg[256];
    __shared__ int   sfar;
    int b = blockIdx.x, tid = threadIdx.x;
    const float* xb = xyz + (size_t)b * N * 3;
    for (int j = tid; j < N; j += 256) dist[j] = 1e10f;
    if (tid == 0) sfar = 0;
    __syncthreads();
    for (int it = 0; it < npoint; ++it) {
        int far = sfar;
        if (tid == 0) out[b * npoint + it] = far;
        float cx = xb[far * 3 + 0], cy = xb[far * 3 + 1], cz = xb[far * 3 + 2];
        float best = -1.0f; int bi = 0;
        for (int j = tid; j < N; j += 256) {
            float dx = xb[j * 3 + 0] - cx;
            float dy = xb[j * 3 + 1] - cy;
            float dz = xb[j * 3 + 2] - cz;
            float nd = fminf(dist[j], dx * dx + dy * dy + dz * dz);
            dist[j] = nd;
            if (nd > best) { best = nd; bi = j; }
        }
        rmax[tid] = best; rarg[tid] = bi;
        __syncthreads();
        for (int off = 128; off > 0; off >>= 1) {
            if (tid < off && rmax[tid + off] > rmax[tid]) {
                rmax[tid] = rmax[tid + off]; rarg[tid] = rarg[tid + off];
            }
            __syncthreads();
        }
        if (tid == 0) sfar = rarg[0];
        __syncthreads();
    }
}

__global__ void k_gather_xyz(const float* __restrict__ xyz, const int* __restrict__ idx,
                             float* __restrict__ out, int N, int S) {
    int i = blockIdx.x * blockDim.x + threadIdx.x;
    if (i >= BB * S * 3) return;
    int b = i / (S * 3);
    int r = i % (S * 3);
    int s = r / 3, c = r % 3;
    out[i] = xyz[((size_t)b * N + idx[b * S + s]) * 3 + c];
}

// ---------------- ball query ----------------
__global__ void k_query_ball(const float* __restrict__ xyz, const float* __restrict__ nxyz,
                             int* __restrict__ idx, int N, int S, int K, float r2) {
    int t = blockIdx.x * blockDim.x + threadIdx.x;
    if (t >= BB * S) return;
    int b = t / S, s = t % S;
    const float* xb = xyz + (size_t)b * N * 3;
    float qx = nxyz[((size_t)b * S + s) * 3 + 0];
    float qy = nxyz[((size_t)b * S + s) * 3 + 1];
    float qz = nxyz[((size_t)b * S + s) * 3 + 2];
    int* ob = idx + (size_t)t * K;
    int cnt = 0, first = 0; bool have = false;
    for (int j = 0; j < N && cnt < K; ++j) {
        float dx = xb[j * 3 + 0] - qx;
        float dy = xb[j * 3 + 1] - qy;
        float dz = xb[j * 3 + 2] - qz;
        if (dx * dx + dy * dy + dz * dz <= r2) {
            if (!have) { first = j; have = true; }
            ob[cnt++] = j;
        }
    }
    for (int c = cnt; c < K; ++c) ob[c] = first;
}

// ---------------- grouping: build (B, C, S*K) with xyz diff + gathered features ----------------
__global__ void k_group(const float* __restrict__ xyz, const float* __restrict__ nxyz,
                        const float* __restrict__ pts, const int* __restrict__ idx,
                        float* __restrict__ Xg, int N, int S, int K, int Cpts, int C) {
    size_t i = (size_t)blockIdx.x * blockDim.x + threadIdx.x;
    size_t total = (size_t)BB * C * S * K;
    if (i >= total) return;
    int k = (int)(i % K);
    size_t r = i / K;
    int s = (int)(r % S); r /= S;
    int c = (int)(r % C);
    int b = (int)(r / C);
    int id = idx[((size_t)b * S + s) * K + k];
    float v;
    if (c < 3)
        v = xyz[((size_t)b * N + id) * 3 + c] - nxyz[((size_t)b * S + s) * 3 + c];
    else
        v = pts[((size_t)b * Cpts + (c - 3)) * N + id];
    Xg[i] = v;
}

// SA3: group-all over N2 points (S = K = N2)
__global__ void k_group_all(const float* __restrict__ xyz2, const float* __restrict__ pts,
                            float* __restrict__ Xg, int N2, int Cpts, int C) {
    size_t i = (size_t)blockIdx.x * blockDim.x + threadIdx.x;
    size_t total = (size_t)BB * C * N2 * N2;
    if (i >= total) return;
    int k = (int)(i % N2);
    size_t r = i / N2;
    int s = (int)(r % N2); r /= N2;
    int c = (int)(r % C);
    int b = (int)(r / C);
    float v;
    if (c < 3)
        v = xyz2[((size_t)b * N2 + k) * 3 + c] - xyz2[((size_t)b * N2 + s) * 3 + c];
    else
        v = pts[((size_t)b * Cpts + (c - 3)) * N2 + k];
    Xg[i] = v;
}

// ---------------- 3-NN + inverse-distance weights ----------------
__global__ void k_three_nn(const float* __restrict__ xyz1, const float* __restrict__ xyz2,
                           int* __restrict__ idx, float* __restrict__ w, int N1, int N2) {
    int t = blockIdx.x * blockDim.x + threadIdx.x;
    if (t >= BB * N1) return;
    int b = t / N1, n = t % N1;
    float px = xyz1[((size_t)b * N1 + n) * 3 + 0];
    float py = xyz1[((size_t)b * N1 + n) * 3 + 1];
    float pz = xyz1[((size_t)b * N1 + n) * 3 + 2];
    const float* xb = xyz2 + (size_t)b * N2 * 3;
    float d0 = 1e30f, d1 = 1e30f, d2 = 1e30f;
    int i0 = 0, i1 = 0, i2 = 0;
    for (int j = 0; j < N2; ++j) {
        float dx = xb[j * 3 + 0] - px;
        float dy = xb[j * 3 + 1] - py;
        float dz = xb[j * 3 + 2] - pz;
        float d = dx * dx + dy * dy + dz * dz;
        if (d < d0)      { d2 = d1; i2 = i1; d1 = d0; i1 = i0; d0 = d; i0 = j; }
        else if (d < d1) { d2 = d1; i2 = i1; d1 = d;  i1 = j; }
        else if (d < d2) { d2 = d;  i2 = j; }
    }
    float w0 = 1.0f / (d0 + 1e-8f);
    float w1 = 1.0f / (d1 + 1e-8f);
    float w2 = 1.0f / (d2 + 1e-8f);
    float ws = w0 + w1 + w2;
    size_t o = (size_t)t * 3;
    idx[o + 0] = i0; idx[o + 1] = i1; idx[o + 2] = i2;
    w[o + 0] = w0 / ws; w[o + 1] = w1 / ws; w[o + 2] = w2 / ws;
}

// x (B, C1+C2, N1): [points1 ; interpolated points2]
__global__ void k_interp_concat(const float* __restrict__ p1, const float* __restrict__ p2,
                                const int* __restrict__ idx, const float* __restrict__ w,
                                float* __restrict__ X, int C1, int C2, int N1, int N2) {
    size_t i = (size_t)blockIdx.x * blockDim.x + threadIdx.x;
    int C = C1 + C2;
    size_t total = (size_t)BB * C * N1;
    if (i >= total) return;
    int n = (int)(i % N1);
    size_t r = i / N1;
    int c = (int)(r % C);
    int b = (int)(r / C);
    float v;
    if (c < C1) {
        v = p1[((size_t)b * C1 + c) * N1 + n];
    } else {
        int cc = c - C1;
        size_t o = ((size_t)b * N1 + n) * 3;
        const float* pb = p2 + ((size_t)b * C2 + cc) * N2;
        v = w[o + 0] * pb[idx[o + 0]] +
            w[o + 1] * pb[idx[o + 1]] +
            w[o + 2] * pb[idx[o + 2]];
    }
    X[i] = v;
}

// ---------------- misc ----------------
__global__ void k_transpose_xt(const float* __restrict__ x_t, float* __restrict__ l0) {
    int i = blockIdx.x * blockDim.x + threadIdx.x;
    if (i >= BB * 3 * NNPTS) return;
    int n = i % NNPTS;
    int r = i / NNPTS;
    int c = r % 3, b = r / 3;
    l0[i] = x_t[((size_t)b * NNPTS + n) * 3 + c];
}

__global__ void k_head_out(const float* __restrict__ X, const float* __restrict__ W2,
                           const float* __restrict__ b2, float* __restrict__ out, int L) {
    size_t i = (size_t)blockIdx.x * blockDim.x + threadIdx.x;
    size_t total = (size_t)BB * 3 * L;
    if (i >= total) return;
    int l = (int)(i % L);
    size_t r = i / L;
    int o = (int)(r % 3);
    int b = (int)(r / 3);
    const float* xb = X + (size_t)b * 128 * L;
    float s = b2[o];
    for (int c = 0; c < 128; ++c) s += W2[o * 128 + c] * xb[(size_t)c * L + l];
    out[i] = s;
}

// ================= host orchestration =================
static inline int nblk(size_t n) { return (int)((n + 255) / 256); }

struct CC { const float *W, *cb, *tmw, *tmb, *g, *be; };

static void run_cc(const CC& p, const float* te, const float* Xin, float* Yout,
                   _Float16* Wh, _Float16* Xh,
                   float* tp, float* b2, float* mean, float* rstd,
                   int Cout, int Cin, int L, hipStream_t s) {
    int Cpad = (Cin + 31) & ~31;
    k_pack_w<<<nblk((size_t)Cout * Cpad), 256, 0, s>>>(p.W, Wh, Cin, Cpad, Cout);
    k_pack_x<<<nblk((size_t)BB * L * Cpad), 256, 0, s>>>(Xin, Xh, Cin, Cpad, L);
    k_tproj<<<nblk((size_t)BB * Cin), 256, 0, s>>>(te, p.tmw, p.tmb, tp, Cin);
    k_bias2<<<nblk((size_t)BB * Cout), 256, 0, s>>>(p.W, p.cb, tp, b2, Cin, Cout);
    dim3 g(L / 16, Cout / 64, BB);
    k_gemm_wmma<<<g, 64, 0, s>>>(Wh, Xh, b2, Yout, Cout, Cpad, L, Cout);
    k_bn_stats<<<Cout, 256, 0, s>>>(Yout, mean, rstd, Cout, L);
    size_t tot = (size_t)BB * Cout * L;
    k_bn_gelu<<<nblk(tot), 256, 0, s>>>(Yout, mean, rstd, p.g, p.be, Cout, L, tot);
}

extern "C" void kernel_launch(void* const* d_in, const int* in_sizes, int n_in,
                              void* d_out, int out_size, void* d_ws, size_t ws_size,
                              hipStream_t stream) {
    (void)in_sizes; (void)n_in; (void)out_size; (void)ws_size;

    const float* x_t    = (const float*)d_in[0];    // (8, 2048, 3)
    const float* t_in   = (const float*)d_in[1];    // (8,)
    const float* time_w = (const float*)d_in[2];
    const float* time_b = (const float*)d_in[3];
    auto cc = [&](int base) -> CC {
        return { (const float*)d_in[base + 0], (const float*)d_in[base + 1],
                 (const float*)d_in[base + 2], (const float*)d_in[base + 3],
                 (const float*)d_in[base + 4], (const float*)d_in[base + 5] };
    };
    CC sa1_0 = cc(4),  sa1_1 = cc(10), sa1_2 = cc(16);
    CC sa2_0 = cc(22), sa2_1 = cc(28), sa2_2 = cc(34);
    CC sa3_0 = cc(40), sa3_1 = cc(46), sa3_2 = cc(52);
    CC fp3_0 = cc(58), fp3_1 = cc(64);
    CC fp2_0 = cc(70), fp2_1 = cc(76);
    CC fp1_0 = cc(82), fp1_1 = cc(88), fp1_2 = cc(94);
    const float* head_w1 = (const float*)d_in[100];
    const float* head_b1 = (const float*)d_in[101];
    const float* head_g  = (const float*)d_in[102];
    const float* head_be = (const float*)d_in[103];
    const float* head_w2 = (const float*)d_in[104];
    const float* head_b2 = (const float*)d_in[105];
    float* out = (float*)d_out;

    // workspace carving (256B aligned)
    char* ws = (char*)d_ws;
    size_t off = 0;
    auto carve = [&](size_t bytes) -> char* {
        char* p = ws + off;
        off += (bytes + 255) & ~(size_t)255;
        return p;
    };
    float*     emb  = (float*)carve((size_t)BB * TD * 4);
    float*     te   = (float*)carve((size_t)BB * TD * 4);
    float*     tp   = (float*)carve((size_t)BB * 1280 * 4);
    float*     b2   = (float*)carve((size_t)BB * 1024 * 4);
    float*     mean = (float*)carve((size_t)1024 * 4);
    float*     rstd = (float*)carve((size_t)1024 * 4);
    int*       fid1 = (int*)carve((size_t)BB * 512 * 4);
    int*       fid2 = (int*)carve((size_t)BB * 128 * 4);
    float*     l1x  = (float*)carve((size_t)BB * 512 * 3 * 4);
    float*     l2x  = (float*)carve((size_t)BB * 128 * 3 * 4);
    int*       qb1  = (int*)carve((size_t)BB * 512 * 32 * 4);
    int*       qb2  = (int*)carve((size_t)BB * 128 * 64 * 4);
    float*     l0   = (float*)carve((size_t)BB * 3 * NNPTS * 4);
    float*     l1   = (float*)carve((size_t)BB * 128 * 512 * 4);
    float*     l2   = (float*)carve((size_t)BB * 256 * 128 * 4);
    float*     l3   = (float*)carve((size_t)BB * 1024 * 128 * 4);
    float*     l2b  = (float*)carve((size_t)BB * 256 * 128 * 4);
    float*     l1b  = (float*)carve((size_t)BB * 128 * 512 * 4);
    int*       nni  = (int*)carve((size_t)BB * NNPTS * 3 * 4);
    float*     nnw  = (float*)carve((size_t)BB * NNPTS * 3 * 4);
    _Float16*  Wh   = (_Float16*)carve((size_t)1024 * 1280 * 2);
    _Float16*  Xh   = (_Float16*)carve((size_t)BB * 16384 * 512 * 2);
    float*     bufA = (float*)carve((size_t)BB * 1024 * 16384 * 4);
    float*     bufB = (float*)carve((size_t)BB * 1024 * 16384 * 4);

    // ---- time embedding ----
    k_time_sincos<<<nblk(BB * TD), 256, 0, stream>>>(t_in, emb);
    k_time_proj<<<nblk(BB * TD), 256, 0, stream>>>(emb, time_w, time_b, te);

    // ---- l0 = transpose(x_t) ----
    k_transpose_xt<<<nblk((size_t)BB * 3 * NNPTS), 256, 0, stream>>>(x_t, l0);

    // ---- SA1: npoint=512, r=0.2, K=32, chain 6->64->64->128 ----
    k_fps<<<BB, 256, 0, stream>>>(x_t, fid1, NNPTS, 512);
    k_gather_xyz<<<nblk((size_t)BB * 512 * 3), 256, 0, stream>>>(x_t, fid1, l1x, NNPTS, 512);
    k_query_ball<<<nblk((size_t)BB * 512), 256, 0, stream>>>(x_t, l1x, qb1, NNPTS, 512, 32, 0.04f);
    k_group<<<nblk((size_t)BB * 6 * 512 * 32), 256, 0, stream>>>(x_t, l1x, l0, qb1, bufA, NNPTS, 512, 32, 3, 6);
    run_cc(sa1_0, te, bufA, bufB, Wh, Xh, tp, b2, mean, rstd, 64, 6, 16384, stream);
    run_cc(sa1_1, te, bufB, bufA, Wh, Xh, tp, b2, mean, rstd, 64, 64, 16384, stream);
    run_cc(sa1_2, te, bufA, bufB, Wh, Xh, tp, b2, mean, rstd, 128, 64, 16384, stream);
    k_maxpool<<<nblk((size_t)BB * 128 * 512), 256, 0, stream>>>(bufB, l1, 128, 512, 32);

    // ---- SA2: npoint=128, r=0.4, K=64, chain 131->128->128->256 ----
    k_fps<<<BB, 256, 0, stream>>>(l1x, fid2, 512, 128);
    k_gather_xyz<<<nblk((size_t)BB * 128 * 3), 256, 0, stream>>>(l1x, fid2, l2x, 512, 128);
    k_query_ball<<<nblk((size_t)BB * 128), 256, 0, stream>>>(l1x, l2x, qb2, 512, 128, 64, 0.16f);
    k_group<<<nblk((size_t)BB * 131 * 128 * 64), 256, 0, stream>>>(l1x, l2x, l1, qb2, bufA, 512, 128, 64, 128, 131);
    run_cc(sa2_0, te, bufA, bufB, Wh, Xh, tp, b2, mean, rstd, 128, 131, 8192, stream);
    run_cc(sa2_1, te, bufB, bufA, Wh, Xh, tp, b2, mean, rstd, 128, 128, 8192, stream);
    run_cc(sa2_2, te, bufA, bufB, Wh, Xh, tp, b2, mean, rstd, 256, 128, 8192, stream);
    k_maxpool<<<nblk((size_t)BB * 256 * 128), 256, 0, stream>>>(bufB, l2, 256, 128, 64);

    // ---- SA3: group-all (S=K=128), chain 259->256->512->1024 ----
    k_group_all<<<nblk((size_t)BB * 259 * 128 * 128), 256, 0, stream>>>(l2x, l2, bufA, 128, 256, 259);
    run_cc(sa3_0, te, bufA, bufB, Wh, Xh, tp, b2, mean, rstd, 256, 259, 16384, stream);
    run_cc(sa3_1, te, bufB, bufA, Wh, Xh, tp, b2, mean, rstd, 512, 256, 16384, stream);
    run_cc(sa3_2, te, bufA, bufB, Wh, Xh, tp, b2, mean, rstd, 1024, 512, 16384, stream);
    k_maxpool<<<nblk((size_t)BB * 1024 * 128), 256, 0, stream>>>(bufB, l3, 1024, 128, 128);

    // ---- FP3: xyz1=l2x (128), xyz2=l2x (SA3 kept same points); 1280->256->256 ----
    k_three_nn<<<nblk((size_t)BB * 128), 256, 0, stream>>>(l2x, l2x, nni, nnw, 128, 128);
    k_interp_concat<<<nblk((size_t)BB * 1280 * 128), 256, 0, stream>>>(l2, l3, nni, nnw, bufA, 256, 1024, 128, 128);
    run_cc(fp3_0, te, bufA, bufB, Wh, Xh, tp, b2, mean, rstd, 256, 1280, 128, stream);
    run_cc(fp3_1, te, bufB, l2b, Wh, Xh, tp, b2, mean, rstd, 256, 256, 128, stream);

    // ---- FP2: xyz1=l1x (512), xyz2=l2x (128); 384->256->128 ----
    k_three_nn<<<nblk((size_t)BB * 512), 256, 0, stream>>>(l1x, l2x, nni, nnw, 512, 128);
    k_interp_concat<<<nblk((size_t)BB * 384 * 512), 256, 0, stream>>>(l1, l2b, nni, nnw, bufA, 128, 256, 512, 128);
    run_cc(fp2_0, te, bufA, bufB, Wh, Xh, tp, b2, mean, rstd, 256, 384, 512, stream);
    run_cc(fp2_1, te, bufB, l1b, Wh, Xh, tp, b2, mean, rstd, 128, 256, 512, stream);

    // ---- FP1: xyz1=x_t (2048), xyz2=l1x (512); 131->128->128->128 ----
    k_three_nn<<<nblk((size_t)BB * NNPTS), 256, 0, stream>>>(x_t, l1x, nni, nnw, NNPTS, 512);
    k_interp_concat<<<nblk((size_t)BB * 131 * NNPTS), 256, 0, stream>>>(l0, l1b, nni, nnw, bufA, 3, 128, NNPTS, 512);
    run_cc(fp1_0, te, bufA, bufB, Wh, Xh, tp, b2, mean, rstd, 128, 131, 2048, stream);
    run_cc(fp1_1, te, bufB, bufA, Wh, Xh, tp, b2, mean, rstd, 128, 128, 2048, stream);
    run_cc(fp1_2, te, bufA, bufB, Wh, Xh, tp, b2, mean, rstd, 128, 128, 2048, stream);

    // ---- head: conv(128->128)+BN+gelu, then conv(128->3) ----
    {
        k_pack_w<<<nblk((size_t)128 * 128), 256, 0, stream>>>(head_w1, Wh, 128, 128, 128);
        k_pack_x<<<nblk((size_t)BB * 2048 * 128), 256, 0, stream>>>(bufB, Xh, 128, 128, 2048);
        dim3 gh(2048 / 16, 128 / 64, BB);
        k_gemm_wmma<<<gh, 64, 0, stream>>>(Wh, Xh, head_b1, bufA, 128, 128, 2048, 0);
        k_bn_stats<<<128, 256, 0, stream>>>(bufA, mean, rstd, 128, 2048);
        size_t th = (size_t)BB * 128 * 2048;
        k_bn_gelu<<<nblk(th), 256, 0, stream>>>(bufA, mean, rstd, head_g, head_be, 128, 2048, th);
        k_head_out<<<nblk((size_t)BB * 3 * NNPTS), 256, 0, stream>>>(bufA, head_w2, head_b2, out, NNPTS);
    }
}
